// StabilityPredictorSchnet_13443247636507
// MI455X (gfx1250) — matrix operand order repaired and checked by reading, not compiled, not run on hardware
//
#include <hip/hip_runtime.h>
#include <hip/hip_bf16.h>

// ---------------------------------------------------------------------------
// StabilityPredictorSchnet for MI455X (gfx1250, wave32, WMMA)
// B=4, L=384, H=128, F=384.  Dominant cost: per-edge 2-layer MLP => two GEMMs
// per (b,l) pair, fused with gelu and the h_V contraction. f16 WMMA, fp32 acc.
// Loop order: kk outer / (jl,i) inner -> 12 WMMA per 4 A-frag + 3 B-frag loads
// (3x less LDS traffic per WMMA than jl-outer order).
// ---------------------------------------------------------------------------

#define B_ 4
#define L_ 384
#define H_ 128
#define F_ 384
#define CH_ 64          // k-rows per chunk (LDS: 16KB A + 48KB T = 64KB)

typedef __attribute__((ext_vector_type(16))) _Float16 v16h;
typedef __attribute__((ext_vector_type(8)))  _Float16 v8h;
typedef __attribute__((ext_vector_type(8)))  float    v8f;

// Exact-erf gelu: used only in the (tiny) head kernels.
__device__ __forceinline__ float geluf(float x) {
    return 0.5f * x * (1.0f + erff(x * 0.70710678118654752440f));
}

// Fast gelu for the f16 edge pipeline: x * sigmoid(1.5957691x + 0.0713548x^3).
// 3 VALU + v_exp_f32 + v_rcp_f32 (TRANS ops co-execute with XDL WMMA).
// Max abs error ~3e-3 == same order as the f16 rounding already applied to T.
__device__ __forceinline__ float fast_gelu(float x) {
    float z = x * (1.59576912f + 0.07135481628f * x * x);
    float e = __expf(-z);                       // v_exp_f32 path
    return x * __builtin_amdgcn_rcpf(1.0f + e); // v_rcp_f32
}

__device__ __forceinline__ v16h frag_cat(v8h lo, v8h hi) {
    return __builtin_shufflevector(lo, hi, 0,1,2,3,4,5,6,7,8,9,10,11,12,13,14,15);
}

// ---------------------------------------------------------------------------
// Pack weight [K,N] fp32 row-major -> f16 fragment layout [K/16][N][16]
// so a B-fragment lane (N=lane&15, K-group by lane>=16) reads 16 contiguous
// halfs (two 16-byte loads).
// ---------------------------------------------------------------------------
__global__ void pack_w_kernel(const float* __restrict__ w, _Float16* __restrict__ out,
                              int K, int N) {
    int idx = blockIdx.x * blockDim.x + threadIdx.x;   // over (K/16)*N
    int total = (K >> 4) * N;
    if (idx >= total) return;
    int kc = idx / N;
    int n  = idx - kc * N;
    _Float16* dst = out + (size_t)idx * 16;
    #pragma unroll
    for (int j = 0; j < 16; ++j)
        dst[j] = (_Float16)w[(size_t)(kc * 16 + j) * N + n];
}

// ---------------------------------------------------------------------------
// Main fused CFConv kernel: one workgroup per (b,l), 256 threads = 8 waves.
// ---------------------------------------------------------------------------
__global__ __launch_bounds__(256) void cfconv_kernel(
    const float*    __restrict__ hV,     // [B,L,F]
    const float*    __restrict__ hE,     // [B,L,L,H]
    const float*    __restrict__ fb1,    // [F]
    const float*    __restrict__ fb2,    // [F]
    const _Float16* __restrict__ fwP1,   // [H/16][F][16]
    const _Float16* __restrict__ fwP2,   // [F/16][F][16]
    float*          __restrict__ xconv)  // [B,L,F]
{
    __shared__ _Float16 Ae[CH_ * H_];    // 16 KB: h_E chunk (f16, row-major)
    __shared__ _Float16 Tt[CH_ * F_];    // 48 KB: gelu(GEMM1) (f16, row-major)

    const int l  = blockIdx.x;
    const int b  = blockIdx.y;
    const int t  = threadIdx.x;
    const int wv = t >> 5;               // wave id 0..7
    const int lane = t & 31;
    const int hi = (lane >> 4) & 1;      // lane-half selects K sub-range
    const int ln = lane & 15;            // N / M within tile

    float xacc[3] = {0.0f, 0.0f, 0.0f};
    const size_t eRow = ((size_t)(b * L_ + l)) * L_;   // h_E row base (rows of H_ floats)

    for (int k0 = 0; k0 < L_; k0 += CH_) {
        // ---- stage h_E chunk fp32 -> f16 into LDS (64 rows x 128 cols) ----
        {
            const int row = t >> 2;
            const int cb  = (t & 3) * 32;
            const float4* src = (const float4*)(hE + (eRow + (size_t)(k0 + row)) * H_ + cb);
            _Float16* dst = &Ae[row * H_ + cb];
            #pragma unroll
            for (int q = 0; q < 8; ++q) {
                float4 v = src[q];
                dst[q * 4 + 0] = (_Float16)v.x;
                dst[q * 4 + 1] = (_Float16)v.y;
                dst[q * 4 + 2] = (_Float16)v.z;
                dst[q * 4 + 3] = (_Float16)v.w;
            }
            // prefetch next chunk's slice (global_prefetch_b8) to hide the
            // HBM latency of the next staging phase behind this chunk's GEMMs
            if (k0 + CH_ < L_) {
                const char* nx = (const char*)(hE + (eRow + (size_t)(k0 + CH_ + row)) * H_ + cb);
                __builtin_prefetch(nx, 0, 1);
                __builtin_prefetch(nx + 64, 0, 1);
            }
        }
        __syncthreads();

        // ---- GEMM1: T = gelu(Ae @ fw1 + fb1), [CH_,F_], K = H_ = 128 ----
        {
            v8f acc[3][4];
            v8f zero = {};
            #pragma unroll
            for (int jl = 0; jl < 3; ++jl)
                #pragma unroll
                for (int i = 0; i < 4; ++i) acc[jl][i] = zero;

            #pragma unroll 1
            for (int kk = 0; kk < 4; ++kk) {          // K steps of 32
                v16h Af[4];
                #pragma unroll
                for (int i = 0; i < 4; ++i) {
                    const _Float16* ap = &Ae[(i * 16 + ln) * H_ + kk * 32 + hi * 16];
                    Af[i] = frag_cat(*(const v8h*)ap, *(const v8h*)(ap + 8));
                }
                const int kc = kk * 2 + hi;
                #pragma unroll
                for (int jl = 0; jl < 3; ++jl) {
                    const int fcol = (wv * 3 + jl) * 16 + ln;
                    const v8h* q = (const v8h*)(fwP1 + ((size_t)kc * F_ + fcol) * 16);
                    v16h Bf = frag_cat(q[0], q[1]);
                    #pragma unroll
                    for (int i = 0; i < 4; ++i)
                        acc[jl][i] = __builtin_amdgcn_wmma_f32_16x16x32_f16(
                            false, Af[i], false, Bf, (short)0, acc[jl][i], false, false);
                }
            }
            #pragma unroll
            for (int jl = 0; jl < 3; ++jl) {
                const int fcol = (wv * 3 + jl) * 16 + ln;
                const float bias = fb1[fcol];
                #pragma unroll
                for (int i = 0; i < 4; ++i) {
                    #pragma unroll
                    for (int r = 0; r < 8; ++r) {
                        float xv = fast_gelu(acc[jl][i][r] + bias);
                        Tt[(i * 16 + hi * 8 + r) * F_ + fcol] = (_Float16)xv;
                    }
                }
            }
        }
        __syncthreads();

        // ---- GEMM2 + epilogue: gelu(T @ fw2 + fb2) * h_V, reduce over k ----
        {
            v8f acc[3][4];
            v8f zero = {};
            #pragma unroll
            for (int jl = 0; jl < 3; ++jl)
                #pragma unroll
                for (int i = 0; i < 4; ++i) acc[jl][i] = zero;

            #pragma unroll 1
            for (int kk = 0; kk < 12; ++kk) {         // K = F_ = 384
                v16h Af[4];
                #pragma unroll
                for (int i = 0; i < 4; ++i) {
                    const _Float16* ap = &Tt[(i * 16 + ln) * F_ + kk * 32 + hi * 16];
                    Af[i] = frag_cat(*(const v8h*)ap, *(const v8h*)(ap + 8));
                }
                const int kc = kk * 2 + hi;
                #pragma unroll
                for (int jl = 0; jl < 3; ++jl) {
                    const int fcol = (wv * 3 + jl) * 16 + ln;
                    const v8h* q = (const v8h*)(fwP2 + ((size_t)kc * F_ + fcol) * 16);
                    v16h Bf = frag_cat(q[0], q[1]);
                    #pragma unroll
                    for (int i = 0; i < 4; ++i)
                        acc[jl][i] = __builtin_amdgcn_wmma_f32_16x16x32_f16(
                            false, Af[i], false, Bf, (short)0, acc[jl][i], false, false);
                }
            }
            #pragma unroll
            for (int jl = 0; jl < 3; ++jl) {
                const int fcol = (wv * 3 + jl) * 16 + ln;
                const float bias = fb2[fcol];
                float xa = xacc[jl];
                #pragma unroll
                for (int i = 0; i < 4; ++i) {
                    // C/D layout: lane holds 8 consecutive rows (hi*8), fixed col
                    const float* vp = hV + ((size_t)b * L_ + (size_t)(k0 + i * 16 + hi * 8)) * F_ + fcol;
                    #pragma unroll
                    for (int r = 0; r < 8; ++r) {
                        float wf = fast_gelu(acc[jl][i][r] + bias);
                        xa += wf * vp[(size_t)r * F_];
                    }
                }
                xacc[jl] = xa;
            }
        }
        __syncthreads();
    }

    // lanes (n, n+16) hold disjoint row-halves of the same column -> pair-sum
    #pragma unroll
    for (int jl = 0; jl < 3; ++jl) {
        float v = xacc[jl];
        v += __shfl_xor(v, 16, 32);
        if (lane < 16)
            xconv[((size_t)(b * L_ + l)) * F_ + (wv * 3 + jl) * 16 + lane] = v;
    }
}

// ---------------------------------------------------------------------------
// Head: y[m,n] = gelu(b[n] + sum_k x[m,k] * w[k,n])   (tiny vs main kernel)
// ---------------------------------------------------------------------------
__global__ void lin_gelu_kernel(const float* __restrict__ x, const float* __restrict__ w,
                                const float* __restrict__ bias, float* __restrict__ y,
                                int M, int K, int N) {
    int idx = blockIdx.x * blockDim.x + threadIdx.x;
    if (idx >= M * N) return;
    int m = idx / N, n = idx - (idx / N) * N;
    float s = bias[n];
    const float* xr = x + (size_t)m * K;
    for (int k = 0; k < K; ++k) s += xr[k] * w[(size_t)k * N + n];
    y[idx] = geluf(s);
}

// Final layer + masked per-batch reduction: out[b] = sum(pred*mask)/sqrt(max(sum(mask),1))
__global__ __launch_bounds__(384) void head_reduce_kernel(
    const float* __restrict__ h2, const float* __restrict__ hw3,
    const float* __restrict__ hb3, const float* __restrict__ mask,
    float* __restrict__ out) {
    const int b = blockIdx.x;
    const int l = threadIdx.x;       // 384
    float s = hb3[0];
    const float* hr = h2 + ((size_t)(b * L_ + l)) * (H_ / 2);
    #pragma unroll 8
    for (int o = 0; o < H_ / 2; ++o) s += hr[o] * hw3[o];
    float mk = mask[b * L_ + l];

    __shared__ float sp[384];
    __shared__ float sm[384];
    sp[l] = s * mk;
    sm[l] = mk;
    __syncthreads();
    if (l < 128) { sp[l] += sp[l + 256]; sm[l] += sm[l + 256]; }
    __syncthreads();
    for (int st = 128; st > 0; st >>= 1) {
        if (l < st) { sp[l] += sp[l + st]; sm[l] += sm[l + st]; }
        __syncthreads();
    }
    if (l == 0) out[b] = sp[0] / sqrtf(fmaxf(sm[0], 1.0f));
}

// ---------------------------------------------------------------------------
// Launch. Workspace layout (bytes):
//   fwP1  f16 [8][384][16]        @ 0        (98304)
//   fwP2  f16 [24][384][16]       @ 98304    (294912)
//   xconv f32 [1536][384]         @ 393216   (2359296)
//   h1    f32 [1536][128]         @ 2752512  (786432)
//   h2    f32 [1536][64]          @ 3538944  (393216)
// total 3932160 bytes.
// ---------------------------------------------------------------------------
extern "C" void kernel_launch(void* const* d_in, const int* in_sizes, int n_in,
                              void* d_out, int out_size, void* d_ws, size_t ws_size,
                              hipStream_t stream) {
    (void)in_sizes; (void)n_in; (void)out_size; (void)ws_size;
    const float* hV   = (const float*)d_in[0];
    const float* hE   = (const float*)d_in[1];
    const float* mask = (const float*)d_in[2];
    const float* fw1  = (const float*)d_in[3];
    const float* fb1  = (const float*)d_in[4];
    const float* fw2  = (const float*)d_in[5];
    const float* fb2  = (const float*)d_in[6];
    const float* hw1  = (const float*)d_in[7];
    const float* hb1  = (const float*)d_in[8];
    const float* hw2  = (const float*)d_in[9];
    const float* hb2  = (const float*)d_in[10];
    const float* hw3  = (const float*)d_in[11];
    const float* hb3  = (const float*)d_in[12];
    float* out = (float*)d_out;

    char* ws = (char*)d_ws;
    _Float16* fwP1 = (_Float16*)(ws + 0);
    _Float16* fwP2 = (_Float16*)(ws + 98304);
    float* xconv   = (float*)(ws + 393216);
    float* h1      = (float*)(ws + 2752512);
    float* h2      = (float*)(ws + 3538944);

    // pack filter weights to WMMA-fragment-friendly f16 layout
    {
        int tot1 = (H_ / 16) * F_;   // 3072
        pack_w_kernel<<<(tot1 + 255) / 256, 256, 0, stream>>>(fw1, fwP1, H_, F_);
        int tot2 = (F_ / 16) * F_;   // 9216
        pack_w_kernel<<<(tot2 + 255) / 256, 256, 0, stream>>>(fw2, fwP2, F_, F_);
    }

    // fused CFConv (all the FLOPs)
    {
        dim3 grid(L_, B_);
        cfconv_kernel<<<grid, 256, 0, stream>>>(hV, hE, fb1, fb2, fwP1, fwP2, xconv);
    }

    // MLP head (negligible cost)
    {
        int m = B_ * L_;
        int tot = m * H_;            // 196608
        lin_gelu_kernel<<<(tot + 255) / 256, 256, 0, stream>>>(xconv, hw1, hb1, h1, m, F_, H_);
        tot = m * (H_ / 2);          // 98304
        lin_gelu_kernel<<<(tot + 255) / 256, 256, 0, stream>>>(h1, hw2, hb2, h2, m, H_, H_ / 2);
        head_reduce_kernel<<<B_, 384, 0, stream>>>(h2, hw3, hb3, mask, out);
    }
}
